// MultiHeadAttention_40372692582597
// MI455X (gfx1250) — matrix operand order repaired and checked by reading, not compiled, not run on hardware
//
#include <hip/hip_runtime.h>
#include <hip/hip_bf16.h>

// ---------------------------------------------------------------------------
// MultiHeadAttention with relative position (music-transformer skew), CDNA5.
// B=2, N=1024, D_IN=1024, H=16, D_H=64, D_OUT=1024.
// rel[i,j] = Q[i] . E_rel[N-1 + j - i]  (skew trick resolved analytically)
// ---------------------------------------------------------------------------

typedef __attribute__((ext_vector_type(16))) _Float16    v16h;
typedef __attribute__((ext_vector_type(8)))  float       v8f;
typedef __attribute__((ext_vector_type(8)))  unsigned int v8u;

#define NB_   1024
#define HB_   16
#define DHB_  64
#define BB_   2
#define DINB_ 1024
#define HDB_  1024   // H*D_H
#define MB_   2048   // B*N
#define RB_   2047   // 2N-1

// per-wave LDS region in attn kernel
#define LDS_STRIDE_ 10752   // 2*2048 (K dbuf) + 2*2048 (V dbuf) + 2048 (T) + 512 (P)
#define LDS_KOFF_   0
#define LDS_VOFF_   4096
#define LDS_TOFF_   8192
#define LDS_POFF_   10240

static __device__ __forceinline__ v8f wmma_f16(v16h a, v16h b, v8f c) {
    return __builtin_amdgcn_wmma_f32_16x16x32_f16(
        /*neg_a=*/false, a, /*neg_b=*/false, b,
        /*c_mod=*/(short)0, c, /*reuse_a=*/false, /*reuse_b=*/false);
}

// A-operand (16x32 f16): row M = lane&15; K = (v>=4?16:0)+(lane>=16?8:0)+2*(v&3)+h
static __device__ __forceinline__ v16h load_a16(const _Float16* A, int lda,
                                                int row, int k, int lane) {
    const _Float16* p = A + (size_t)row * lda + k;
    const int hi8 = (lane >= 16) ? 8 : 0;
    v8u u;
#pragma unroll
    for (int v = 0; v < 8; ++v) {
        int koff = ((v >= 4) ? 16 : 0) + hi8 + 2 * (v & 3);
        u[v] = *(const unsigned int*)(p + koff);
    }
    return __builtin_bit_cast(v16h, u);
}

// B-operand (32x16 f16): col N = lane&15; K = (v>=4?16:0)+(lane>=16?8:0)+2*(v&3)+h
// from K-contiguous ("Bt") rows: works on global or LDS pointers.
static __device__ __forceinline__ v16h load_bt16(const _Float16* Bt, int ldb,
                                                 int col, int k, int lane) {
    const _Float16* p = Bt + (size_t)col * ldb + k + ((lane >= 16) ? 8 : 0);
    v8u u;
#pragma unroll
    for (int v = 0; v < 8; ++v) u[v] = *(const unsigned int*)(p + ((v >= 4) ? 16 : 0) + 2 * (v & 3));
    return __builtin_bit_cast(v16h, u);
}

// CDNA5 async global->LDS copy (ASYNCcnt tracked): each lane moves 16 bytes.
static __device__ __forceinline__ void async_copy_b128(unsigned lds_byte_off,
                                                       const void* gaddr) {
    unsigned long long ga = (unsigned long long)gaddr;
    asm volatile("global_load_async_to_lds_b128 %0, %1, off"
                 :: "v"(lds_byte_off), "v"(ga) : "memory");
}
static __device__ __forceinline__ void wait_async0() {
    asm volatile("s_wait_asynccnt 0x0" ::: "memory");
}
static __device__ __forceinline__ void wait_async8() {
    asm volatile("s_wait_asynccnt 0x8" ::: "memory");
}

// 32x64 output tile per wave: two A row-tiles share every B operand.
static __device__ __forceinline__ void gemm32x64(const _Float16* A, int lda, int m0,
                                                 const _Float16* Bt, int ldb, int n0,
                                                 int K, int lane, v8f acc[8]) {
    const int dj = lane & 15;
    for (int k = 0; k < K; k += 32) {
        v16h a0 = load_a16(A, lda, m0 + dj, k, lane);
        v16h a1 = load_a16(A, lda, m0 + 16 + dj, k, lane);
#pragma unroll
        for (int nb = 0; nb < 4; ++nb) {
            v16h b = load_bt16(Bt, ldb, n0 + nb * 16 + dj, k, lane);
            acc[nb]     = wmma_f16(a0, b, acc[nb]);
            acc[4 + nb] = wmma_f16(a1, b, acc[4 + nb]);
        }
    }
}

// ---------------------------------------------------------------------------
// Conversion / transpose kernels
// ---------------------------------------------------------------------------
__global__ __launch_bounds__(256) void cvt_f16_kernel(const float* __restrict__ src,
                                                      _Float16* __restrict__ dst, int n) {
    int i = blockIdx.x * blockDim.x + threadIdx.x;
    if (i < n) dst[i] = (_Float16)src[i];
}

__global__ __launch_bounds__(256) void transpose_f16_kernel(const float* __restrict__ W,
                                                            _Float16* __restrict__ Wt,
                                                            int K, int Nn) {
    int idx = blockIdx.x * blockDim.x + threadIdx.x;
    if (idx < K * Nn) {
        int k = idx / Nn, n = idx % Nn;
        Wt[(size_t)n * K + k] = (_Float16)W[idx];
    }
}

// ---------------------------------------------------------------------------
// QKV projection: xh[2048,1024] @ W{q,k,v}T. Wave tile = 32 rows x 64 cols.
// Q/K head-major [b,h,n,d]; V stored transposed [b,h,d,n].
// ---------------------------------------------------------------------------
__global__ __launch_bounds__(128) void qkv_kernel(const _Float16* __restrict__ xh,
                                                  const _Float16* __restrict__ WqT,
                                                  const _Float16* __restrict__ WkT,
                                                  const _Float16* __restrict__ WvT,
                                                  _Float16* __restrict__ Qh,
                                                  _Float16* __restrict__ Kh,
                                                  _Float16* __restrict__ VhT) {
    const int lane = threadIdx.x & 31;
    const int wid = blockIdx.x * 4 + (threadIdx.x >> 5);
    const int mat = wid / (64 * 16);       // 0..2
    const int rem = wid % (64 * 16);
    const int mt = rem / 16;               // 32-row tile (0..63)
    const int ct = rem % 16;               // head index
    const _Float16* Wt = (mat == 0) ? WqT : ((mat == 1) ? WkT : WvT);

    v8f acc[8];
#pragma unroll
    for (int t = 0; t < 8; ++t)
#pragma unroll
        for (int r = 0; r < 8; ++r) acc[t][r] = 0.0f;

    gemm32x64(xh, DINB_, mt * 32, Wt, DINB_, ct * 64, DINB_, lane, acc);

    const int dj = lane & 15;
    const int hi8 = (lane >= 16) ? 8 : 0;
#pragma unroll
    for (int g = 0; g < 2; ++g) {
#pragma unroll
        for (int nb = 0; nb < 4; ++nb) {
#pragma unroll
            for (int r = 0; r < 8; ++r) {
                int m = mt * 32 + g * 16 + r + hi8;
                int bb = m >> 10, nn = m & (NB_ - 1);
                int d = nb * 16 + dj;
                _Float16 val = (_Float16)acc[g * 4 + nb][r];
                if (mat == 2) {
                    VhT[(((size_t)(bb * HB_ + ct)) * DHB_ + d) * NB_ + nn] = val;
                } else {
                    _Float16* dst = (mat == 0) ? Qh : Kh;
                    dst[(((size_t)(bb * HB_ + ct)) * NB_ + nn) * DHB_ + d] = val;
                }
            }
        }
    }
}

// ---------------------------------------------------------------------------
// Fused flash-style attention with relative positions. One wave per
// (b, h, 16-row i-tile). K/V^T j-tiles double-buffered in LDS via
// global_load_async_to_lds_b128; per tile: 2 WMMA (QK^T) + 4 WMMA (rel aux
// T 16x32) + 4 WMMA (PV). Online softmax via wave32 shfl_xor.
// ---------------------------------------------------------------------------
__global__ __launch_bounds__(128) void attn_kernel(const _Float16* __restrict__ Qh,
                                                   const _Float16* __restrict__ Kh,
                                                   const _Float16* __restrict__ VhT,
                                                   const _Float16* __restrict__ Eh,
                                                   _Float16* __restrict__ Ah) {
    __shared__ unsigned char smem[4 * LDS_STRIDE_];
    const int lane  = threadIdx.x & 31;
    const int wslot = threadIdx.x >> 5;
    const int wid   = blockIdx.x * 4 + wslot;
    const int it = wid & 63;
    const int h  = (wid >> 6) & 15;
    const int b  = wid >> 10;
    const int i0 = it * 16;

    const unsigned ldsbase = (unsigned)(wslot * LDS_STRIDE_);   // smem is at LDS offset 0
    float*    Tlds = (float*)(smem + wslot * LDS_STRIDE_ + LDS_TOFF_);
    _Float16* Plds = (_Float16*)(smem + wslot * LDS_STRIDE_ + LDS_POFF_);

    const _Float16* Qp = Qh + (size_t)(b * HB_ + h) * NB_ * DHB_;   // [n][d]
    const _Float16* Kp = Kh + (size_t)(b * HB_ + h) * NB_ * DHB_;   // [n][d]
    const _Float16* Vt = VhT + (size_t)(b * HB_ + h) * DHB_ * NB_;  // [d][n]

    const int arow = i0 + (lane & 15);
    const v16h aq0 = load_a16(Qp, DHB_, arow, 0, lane);
    const v16h aq1 = load_a16(Qp, DHB_, arow, 32, lane);

    // async-stage one (K tile, V^T tile) pair into LDS buffer `buf`
    auto issue_tile = [&](int jt, int buf) {
        const int j0 = jt * 16;
        const unsigned kb = ldsbase + LDS_KOFF_ + (unsigned)buf * 2048u;
        const unsigned vb = ldsbase + LDS_VOFF_ + (unsigned)buf * 2048u;
        const _Float16* gk = Kp + (size_t)j0 * DHB_;     // 16 rows x 64 halves, contiguous
#pragma unroll
        for (int i = 0; i < 4; ++i)
            async_copy_b128(kb + (unsigned)(i * 512 + lane * 16), gk + i * 256 + lane * 8);
#pragma unroll
        for (int i = 0; i < 4; ++i) {                    // V^T rows d, 32B each
            int d = i * 16 + (lane >> 1);
            async_copy_b128(vb + (unsigned)(i * 512 + lane * 16),
                            Vt + (size_t)d * NB_ + j0 + (lane & 1) * 8);
        }
    };

    float mrun[8], lrun[8];
    v8f acc[4];
#pragma unroll
    for (int r = 0; r < 8; ++r) { mrun[r] = -3.0e38f; lrun[r] = 0.0f; }
#pragma unroll
    for (int nb = 0; nb < 4; ++nb)
#pragma unroll
        for (int r = 0; r < 8; ++r) acc[nb][r] = 0.0f;

    const float scale = 0.125f;           // 1/sqrt(64)
    const int hi8 = (lane >= 16) ? 8 : 0;
    const int dj = lane & 15;

    issue_tile(0, 0);

    for (int jt = 0; jt <= it; ++jt) {
        const int j0 = jt * 16;
        const int buf = jt & 1;
        if (jt < it) { issue_tile(jt + 1, buf ^ 1); wait_async8(); }
        else         { wait_async0(); }

        const _Float16* kbuf = (const _Float16*)(smem + wslot * LDS_STRIDE_ + LDS_KOFF_ + buf * 2048);
        const _Float16* vbuf = (const _Float16*)(smem + wslot * LDS_STRIDE_ + LDS_VOFF_ + buf * 2048);

        // ---- S = Q K^T (16x16, K=64), K tile from LDS ----
        v8f s;
#pragma unroll
        for (int r = 0; r < 8; ++r) s[r] = 0.0f;
        s = wmma_f16(aq0, load_bt16(kbuf, DHB_, dj, 0, lane), s);
        s = wmma_f16(aq1, load_bt16(kbuf, DHB_, dj, 32, lane), s);

        // ---- rel aux T[di,kk] = Q[i0+di] . E[ebase+kk], kk in [0,32) ----
        const int ebase = (NB_ - 16) + j0 - i0;          // rows always in [0,2046]
        __builtin_prefetch(Eh + (size_t)(ebase + 32) * DHB_, 0, 1);
        v8f t0, t1;
#pragma unroll
        for (int r = 0; r < 8; ++r) { t0[r] = 0.0f; t1[r] = 0.0f; }
        t0 = wmma_f16(aq0, load_bt16(Eh, DHB_, ebase + dj, 0, lane), t0);
        t0 = wmma_f16(aq1, load_bt16(Eh, DHB_, ebase + dj, 32, lane), t0);
        t1 = wmma_f16(aq0, load_bt16(Eh, DHB_, ebase + 16 + dj, 0, lane), t1);
        t1 = wmma_f16(aq1, load_bt16(Eh, DHB_, ebase + 16 + dj, 32, lane), t1);

        // stage T through LDS to gather the skewed diagonal
#pragma unroll
        for (int r = 0; r < 8; ++r) {
            Tlds[(r + hi8) * 32 + dj]      = t0[r];
            Tlds[(r + hi8) * 32 + 16 + dj] = t1[r];
        }
        asm volatile("s_wait_dscnt 0" ::: "memory");

        // ---- combine + online softmax (row M = r+hi8 lives on 16 lanes) ----
#pragma unroll
        for (int r = 0; r < 8; ++r) {
            const int di = r + hi8;
            const int kk = dj - di + 15;                 // in [0,30]
            float sv = (s[r] + Tlds[di * 32 + kk]) * scale;
            if (jt == it && dj > di) sv += -1.0e9f;      // causal mask, post-scale

            float rmax = sv;
            rmax = fmaxf(rmax, __shfl_xor(rmax, 1, 32));
            rmax = fmaxf(rmax, __shfl_xor(rmax, 2, 32));
            rmax = fmaxf(rmax, __shfl_xor(rmax, 4, 32));
            rmax = fmaxf(rmax, __shfl_xor(rmax, 8, 32));

            float mnew  = fmaxf(mrun[r], rmax);
            float alpha = __expf(mrun[r] - mnew);
            float p     = __expf(sv - mnew);
            float psum  = p;
            psum += __shfl_xor(psum, 1, 32);
            psum += __shfl_xor(psum, 2, 32);
            psum += __shfl_xor(psum, 4, 32);
            psum += __shfl_xor(psum, 8, 32);

            lrun[r] = lrun[r] * alpha + psum;
            mrun[r] = mnew;
#pragma unroll
            for (int nb = 0; nb < 4; ++nb) acc[nb][r] *= alpha;
            Plds[di * 16 + dj] = (_Float16)p;
        }
        asm volatile("s_wait_dscnt 0" ::: "memory");

        // ---- P as A-operand (16 real K, zero-padded to 32) ----
        v8u up;
        {
            const int prow = lane & 15;
            const int kb2 = (lane >= 16) ? 8 : 0;
#pragma unroll
            for (int v = 0; v < 4; ++v)
                up[v] = *(const unsigned int*)(Plds + prow * 16 + kb2 + 2 * v);
#pragma unroll
            for (int v = 4; v < 8; ++v) up[v] = 0u;
        }
        const v16h ap = __builtin_bit_cast(v16h, up);

        // ---- O += P @ V  (V^T tile from LDS; K>=16 garbage nulled by zero A) ----
#pragma unroll
        for (int nb = 0; nb < 4; ++nb) {
            v16h bv = load_bt16(vbuf, 16, nb * 16 + dj, 0, lane);
            acc[nb] = wmma_f16(ap, bv, acc[nb]);
        }
    }

    // ---- normalize and scatter to row-major [b*N+n][h*64+d] (f16) ----
#pragma unroll
    for (int r = 0; r < 8; ++r) {
        const int di = r + hi8;
        const float inv = 1.0f / lrun[r];
        const int m = b * NB_ + i0 + di;
#pragma unroll
        for (int nb = 0; nb < 4; ++nb) {
            const int c = h * 64 + nb * 16 + dj;
            Ah[(size_t)m * HDB_ + c] = (_Float16)(acc[nb][r] * inv);
        }
    }
}

// ---------------------------------------------------------------------------
// Output projection: Ah[2048,1024] @ WoT + bo -> out f32. 32x64 wave tiles.
// ---------------------------------------------------------------------------
__global__ __launch_bounds__(128) void outproj_kernel(const _Float16* __restrict__ Ah,
                                                      const _Float16* __restrict__ WoT,
                                                      const float* __restrict__ bo,
                                                      float* __restrict__ out) {
    const int lane = threadIdx.x & 31;
    const int wid = blockIdx.x * 4 + (threadIdx.x >> 5);
    const int mt = wid / 16;       // 32-row tile (0..63)
    const int ct = wid % 16;

    v8f acc[8];
#pragma unroll
    for (int t = 0; t < 8; ++t)
#pragma unroll
        for (int r = 0; r < 8; ++r) acc[t][r] = 0.0f;

    gemm32x64(Ah, HDB_, mt * 32, WoT, HDB_, ct * 64, HDB_, lane, acc);

    const int dj = lane & 15;
    const int hi8 = (lane >= 16) ? 8 : 0;
#pragma unroll
    for (int g = 0; g < 2; ++g) {
#pragma unroll
        for (int nb = 0; nb < 4; ++nb) {
#pragma unroll
            for (int r = 0; r < 8; ++r) {
                int m = mt * 32 + g * 16 + r + hi8;
                int c = ct * 64 + nb * 16 + dj;
                out[(size_t)m * 1024 + c] = acc[g * 4 + nb][r] + bo[c];
            }
        }
    }
}

// ---------------------------------------------------------------------------
// Host launcher
// ---------------------------------------------------------------------------
extern "C" void kernel_launch(void* const* d_in, const int* in_sizes, int n_in,
                              void* d_out, int out_size, void* d_ws, size_t ws_size,
                              hipStream_t stream) {
    const float* x    = (const float*)d_in[0];   // [2,1024,1024]
    const float* Erel = (const float*)d_in[1];   // [2047,64]
    // d_in[2] = mask (causal mask regenerated analytically)
    const float* Wq   = (const float*)d_in[3];
    const float* Wk   = (const float*)d_in[4];
    const float* Wv   = (const float*)d_in[5];
    const float* Wo   = (const float*)d_in[6];
    const float* bo   = (const float*)d_in[7];
    float* out = (float*)d_out;

    size_t off = 0;
    auto alloc = [&](size_t bytes) -> char* {
        char* p = (char*)d_ws + off;
        off = (off + bytes + 4095) & ~(size_t)255;
        return p;
    };
    _Float16* xh  = (_Float16*)alloc((size_t)MB_ * DINB_ * 2);
    _Float16* WqT = (_Float16*)alloc((size_t)DINB_ * HDB_ * 2);
    _Float16* WkT = (_Float16*)alloc((size_t)DINB_ * HDB_ * 2);
    _Float16* WvT = (_Float16*)alloc((size_t)DINB_ * HDB_ * 2);
    _Float16* WoT = (_Float16*)alloc((size_t)HDB_ * 1024 * 2);
    _Float16* Eh  = (_Float16*)alloc((size_t)RB_ * DHB_ * 2);
    _Float16* Qh  = (_Float16*)alloc((size_t)BB_ * HB_ * NB_ * DHB_ * 2);
    _Float16* Kh  = (_Float16*)alloc((size_t)BB_ * HB_ * NB_ * DHB_ * 2);
    _Float16* VhT = (_Float16*)alloc((size_t)BB_ * HB_ * DHB_ * NB_ * 2);
    _Float16* Ah  = (_Float16*)alloc((size_t)MB_ * HDB_ * 2);
    (void)ws_size; (void)in_sizes; (void)n_in; (void)out_size;

    {
        int n = MB_ * DINB_;
        cvt_f16_kernel<<<(n + 255) / 256, 256, 0, stream>>>(x, xh, n);
    }
    {
        int n = RB_ * DHB_;
        cvt_f16_kernel<<<(n + 255) / 256, 256, 0, stream>>>(Erel, Eh, n);
    }
    {
        int n = DINB_ * HDB_;
        int g = (n + 255) / 256;
        transpose_f16_kernel<<<g, 256, 0, stream>>>(Wq, WqT, DINB_, HDB_);
        transpose_f16_kernel<<<g, 256, 0, stream>>>(Wk, WkT, DINB_, HDB_);
        transpose_f16_kernel<<<g, 256, 0, stream>>>(Wv, WvT, DINB_, HDB_);
        transpose_f16_kernel<<<g, 256, 0, stream>>>(Wo, WoT, HDB_, 1024);
    }

    // QKV projections: 3 mats * 64 row-tiles(32) * 16 heads = 3072 wave-tiles
    qkv_kernel<<<3072 / 4, 128, 0, stream>>>(xh, WqT, WkT, WvT, Qh, Kh, VhT);

    // Fused causal attention + relative positions: 2048 wave-tiles
    attn_kernel<<<2048 / 4, 128, 0, stream>>>(Qh, Kh, VhT, Eh, Ah);

    // Output projection + bias: 64*16 = 1024 wave-tiles
    outproj_kernel<<<1024 / 4, 128, 0, stream>>>(Ah, WoT, bo, out);
}
